// PolyhedralGraphGNN_79534204388001
// MI455X (gfx1250) — matrix operand order repaired and checked by reading, not compiled, not run on hardware
//
#include <hip/hip_runtime.h>
#include <hip/hip_bf16.h>

// ---------------- problem constants (match reference) ----------------
#define N_NODES   100000
#define N_EDGES   640000
#define IN_DIM    64
#define HID       128
#define OUT_DIM   32
#define EDGE_DIM  4
#define N_LAYERS  3
#define NUM_GRAPHS 64
#define LN_EPS    1e-5f

#define MSG_K     (2 * HID + EDGE_DIM)   // 260
#define MSG_KPAD  288                    // 9 K-tiles of 32
#define UPD_KPAD  256                    // 8 K-tiles

// ---------------- WMMA types ----------------
typedef __bf16 bf16_t;
typedef bf16_t bf16x16 __attribute__((ext_vector_type(16)));
typedef float  floatx8 __attribute__((ext_vector_type(8)));

union frag_t { uint4 q[2]; bf16x16 v; };

__device__ __forceinline__ unsigned short f32_to_bf16(float f) {
  unsigned int u = __float_as_uint(f);
  u += 0x7FFFu + ((u >> 16) & 1u);           // round to nearest even
  return (unsigned short)(u >> 16);
}

__device__ __forceinline__ uint2 pack4_bf16(float4 f) {
  uint2 r;
  r.x = (unsigned)f32_to_bf16(f.x) | ((unsigned)f32_to_bf16(f.y) << 16);
  r.y = (unsigned)f32_to_bf16(f.z) | ((unsigned)f32_to_bf16(f.w) << 16);
  return r;
}

// One 16x16 f32 tile of A[32-row LDS, bf16] x Bpacked, accumulating over NK K-tiles.
// A layout (16-bit, 16x32): lanes 0-15 row M=lane, halves = K{0..15};
// lanes 16-31 row M=lane-16, halves = K{16..31}.
// Bpacked: tile (kt*nTilesN+nt), lane L holds column (nt*16 + L&15),
// halves = K{kt*32 + (L>=16?16:0) + 0..15}  -> contiguous 32B per lane.
template <int NK>
__device__ __forceinline__ floatx8 tile_gemm(
    const unsigned short* __restrict__ sA, int ldA, int m0, int lane,
    const unsigned short* __restrict__ wp, int nTilesN, int ntile) {
  floatx8 acc = {0.f, 0.f, 0.f, 0.f, 0.f, 0.f, 0.f, 0.f};
  const int sel = (lane >> 4) & 1;
  const unsigned short* arow = sA + (m0 + (lane & 15)) * ldA;
#pragma unroll
  for (int kt = 0; kt < NK; ++kt) {
    frag_t A, B;
    A.q[0] = *(const uint4*)(arow + kt * 32 + sel * 16);
    A.q[1] = *(const uint4*)(arow + kt * 32 + sel * 16 + 8);
    const unsigned short* bp = wp + ((kt * nTilesN + ntile) * 32 + lane) * 16;
    B.q[0] = *(const uint4*)(bp);
    B.q[1] = *(const uint4*)(bp + 8);
    acc = __builtin_amdgcn_wmma_f32_16x16x32_bf16(
        false, A.v, false, B.v, (short)0, acc, false, false);
  }
  return acc;
}

// ---------------- weight packer: f32 [K,N] row-major -> bf16 B-fragment tiles --------
__global__ __launch_bounds__(256) void pack_w_kernel(
    const float* __restrict__ src, int K, int N, int nKtiles,
    unsigned short* __restrict__ dst) {
  int gid  = blockIdx.x * blockDim.x + threadIdx.x;
  int lane = gid & 31;
  int tile = gid >> 5;
  int nTilesN = N >> 4;
  if (tile >= nKtiles * nTilesN) return;
  int kt = tile / nTilesN;
  int nt = tile - kt * nTilesN;
  int kbase = kt * 32 + ((lane >> 4) << 4);
  int colN  = nt * 16 + (lane & 15);
  unsigned short* d = dst + ((size_t)tile * 32 + lane) * 16;
#pragma unroll
  for (int kk = 0; kk < 16; ++kk) {
    int kidx = kbase + kk;
    float v = (kidx < K) ? src[(size_t)kidx * N + colN] : 0.f;
    d[kk] = f32_to_bf16(v);
  }
}

// ---------------- fill ----------------
__global__ void fill_zero_kernel(float* __restrict__ p, int n) {
  int i = blockIdx.x * blockDim.x + threadIdx.x;
  if (i < n) p[i] = 0.f;
}

// ---------------- input projection: h = relu(x @ in_w + b) ----------------
__global__ __launch_bounds__(256) void in_proj_kernel(
    const float* __restrict__ x, const unsigned short* __restrict__ wp,
    const float* __restrict__ b, float* __restrict__ h) {
  __shared__ alignas(16) unsigned short sA[32 * IN_DIM];
  const int tid = threadIdx.x;
  const int n0  = blockIdx.x * 32;
  {  // 32 rows x 64 halves, float4 loads + uint2 packed stores
    int m  = tid >> 3;
    int c0 = (tid & 7) * 2;  // chunks of 4 floats, 16 chunks per row
    const float4* xp = (const float4*)(x + (size_t)(n0 + m) * IN_DIM);
    uint2* dp = (uint2*)(sA + m * IN_DIM);
    dp[c0]     = pack4_bf16(xp[c0]);
    dp[c0 + 1] = pack4_bf16(xp[c0 + 1]);
  }
  __syncthreads();
  const int lane = tid & 31, wv = tid >> 5, sel = lane >> 4;
  const int cN = wv * 16 + (lane & 15);
  const float bias = b[cN];
#pragma unroll
  for (int mt = 0; mt < 2; ++mt) {
    floatx8 acc = tile_gemm<2>(sA, IN_DIM, mt * 16, lane, wp, 8, wv);
#pragma unroll
    for (int r = 0; r < 8; ++r) {
      float v = acc[r] + bias;
      v = v > 0.f ? v : 0.f;
      h[(size_t)(n0 + mt * 16 + r + sel * 8) * HID + cN] = v;
    }
  }
}

// ------- fused edge kernel: gather -> MLP(260->128 relu ->128) -> scatter-add -------
__global__ __launch_bounds__(256) void edge_msg_kernel(
    const float* __restrict__ h, const float* __restrict__ edge_attr,
    const int* __restrict__ erow, const int* __restrict__ ecol,
    const unsigned short* __restrict__ w1p, const float* __restrict__ b1,
    const unsigned short* __restrict__ w2p, const float* __restrict__ b2,
    float* __restrict__ aggr) {
  __shared__ alignas(16) unsigned short sA[32 * MSG_KPAD];
  __shared__ alignas(16) unsigned short sM[32 * HID];
  const int tid = threadIdx.x;
  const int e0  = blockIdx.x * 32;
  {  // build A = [h[row] | h[col] | edge_attr | 0-pad], 72 chunks of 4 per row
    int m  = tid >> 3;
    int k0 = (tid & 7) * 9;   // 9 chunks each
    int e  = e0 + m;
    const float4* hr = (const float4*)(h + (size_t)erow[e] * HID);  // 32 chunks
    const float4* hc = (const float4*)(h + (size_t)ecol[e] * HID);  // 32 chunks
    uint2* dp = (uint2*)(sA + m * MSG_KPAD);
    for (int c = k0; c < k0 + 9; ++c) {
      float4 f;
      if (c < 32)       f = hr[c];
      else if (c < 64)  f = hc[c - 32];
      else if (c == 64) f = *(const float4*)(edge_attr + (size_t)e * EDGE_DIM);
      else              f = make_float4(0.f, 0.f, 0.f, 0.f);
      dp[c] = pack4_bf16(f);
    }
  }
  __syncthreads();
  const int lane = tid & 31, wv = tid >> 5, sel = lane >> 4;
  const int cN = wv * 16 + (lane & 15);
  const float bias1 = b1[cN];
#pragma unroll
  for (int mt = 0; mt < 2; ++mt) {
    floatx8 acc = tile_gemm<MSG_KPAD / 32>(sA, MSG_KPAD, mt * 16, lane, w1p, 8, wv);
#pragma unroll
    for (int r = 0; r < 8; ++r) {
      float v = acc[r] + bias1;
      v = v > 0.f ? v : 0.f;
      sM[(mt * 16 + r + sel * 8) * HID + cN] = f32_to_bf16(v);
    }
  }
  __syncthreads();
  const float bias2 = b2[cN];
#pragma unroll
  for (int mt = 0; mt < 2; ++mt) {
    floatx8 acc = tile_gemm<HID / 32>(sM, HID, mt * 16, lane, w2p, 8, wv);
#pragma unroll
    for (int r = 0; r < 8; ++r) {
      int m  = mt * 16 + r + sel * 8;
      int dn = erow[e0 + m];
      atomicAdd(&aggr[(size_t)dn * HID + cN], acc[r] + bias2);
    }
  }
}

// ------- fused node update: MLP(256->128 relu ->128) + LayerNorm + relu (+res) -------
__global__ __launch_bounds__(256) void node_upd_kernel(
    float* __restrict__ h, const float* __restrict__ aggr,
    const unsigned short* __restrict__ w1p, const float* __restrict__ b1,
    const unsigned short* __restrict__ w2p, const float* __restrict__ b2,
    const float* __restrict__ lng, const float* __restrict__ lnb,
    int has_residual) {
  __shared__ alignas(16) unsigned short sA[32 * UPD_KPAD];
  __shared__ alignas(16) unsigned short sM[32 * HID];
  __shared__ alignas(16) float sU[32 * HID];
  __shared__ float sPS[256], sPS2[256];
  __shared__ float sMu[32], sRs[32];
  const int tid = threadIdx.x;
  const int n0  = blockIdx.x * 32;
  {  // A = [h | aggr], 64 chunks of 4 per row, float4 loads
    int m  = tid >> 3;
    int c0 = (tid & 7) * 8;   // 8 chunks each
    const float4* hp = (const float4*)(h    + (size_t)(n0 + m) * HID);  // 32 chunks
    const float4* ap = (const float4*)(aggr + (size_t)(n0 + m) * HID);  // 32 chunks
    uint2* dp = (uint2*)(sA + m * UPD_KPAD);
#pragma unroll
    for (int c = c0; c < c0 + 8; ++c)
      dp[c] = pack4_bf16(c < 32 ? hp[c] : ap[c - 32]);
  }
  __syncthreads();
  const int lane = tid & 31, wv = tid >> 5, sel = lane >> 4;
  const int cN = wv * 16 + (lane & 15);
  const float bias1 = b1[cN];
#pragma unroll
  for (int mt = 0; mt < 2; ++mt) {
    floatx8 acc = tile_gemm<UPD_KPAD / 32>(sA, UPD_KPAD, mt * 16, lane, w1p, 8, wv);
#pragma unroll
    for (int r = 0; r < 8; ++r) {
      float v = acc[r] + bias1;
      v = v > 0.f ? v : 0.f;
      sM[(mt * 16 + r + sel * 8) * HID + cN] = f32_to_bf16(v);
    }
  }
  __syncthreads();
  const float bias2 = b2[cN];
#pragma unroll
  for (int mt = 0; mt < 2; ++mt) {
    floatx8 acc = tile_gemm<HID / 32>(sM, HID, mt * 16, lane, w2p, 8, wv);
#pragma unroll
    for (int r = 0; r < 8; ++r)
      sU[(mt * 16 + r + sel * 8) * HID + cN] = acc[r] + bias2;
  }
  __syncthreads();
  {  // LN partials: 8 threads per row, 16 elems each
    int m  = tid >> 3;
    int c0 = (tid & 7) * 16;
    float s = 0.f, s2 = 0.f;
#pragma unroll
    for (int k = 0; k < 16; ++k) {
      float v = sU[m * HID + c0 + k];
      s += v; s2 += v * v;
    }
    sPS[tid] = s; sPS2[tid] = s2;
  }
  __syncthreads();
  if (tid < 32) {  // reduce 8 partials per row
    float s = 0.f, s2 = 0.f;
#pragma unroll
    for (int j = 0; j < 8; ++j) { s += sPS[tid * 8 + j]; s2 += sPS2[tid * 8 + j]; }
    float mu  = s * (1.f / HID);
    float var = s2 * (1.f / HID) - mu * mu;
    sMu[tid] = mu;
    sRs[tid] = rsqrtf(var + LN_EPS);
  }
  __syncthreads();
  {  // apply LN + relu (+residual), float4 write-back
    int m  = tid >> 3;
    int c0 = (tid & 7) * 16;
    float mu = sMu[m], rs = sRs[m];
    float* hp = h + (size_t)(n0 + m) * HID;
#pragma unroll
    for (int q = 0; q < 4; ++q) {
      int c = c0 + q * 4;
      float4 u  = *(const float4*)&sU[m * HID + c];
      float4 gg = *(const float4*)&lng[c];
      float4 bb = *(const float4*)&lnb[c];
      float4 o;
      o.x = fmaxf((u.x - mu) * rs * gg.x + bb.x, 0.f);
      o.y = fmaxf((u.y - mu) * rs * gg.y + bb.y, 0.f);
      o.z = fmaxf((u.z - mu) * rs * gg.z + bb.z, 0.f);
      o.w = fmaxf((u.w - mu) * rs * gg.w + bb.w, 0.f);
      if (has_residual) {
        float4 ho = *(const float4*)&hp[c];
        o.x += ho.x; o.y += ho.y; o.z += ho.z; o.w += ho.w;
      }
      *(float4*)&hp[c] = o;
    }
  }
}

// ---------------- pooling ----------------
__global__ __launch_bounds__(256) void pool_scatter_kernel(
    const float* __restrict__ h, const int* __restrict__ batch,
    float* __restrict__ sums, float* __restrict__ counts) {
  int i = blockIdx.x * blockDim.x + threadIdx.x;
  if (i >= N_NODES * HID) return;
  int node = i >> 7, c = i & (HID - 1);
  int g = batch[node];
  atomicAdd(&sums[g * HID + c], h[i]);
  if (c == 0) atomicAdd(&counts[g], 1.0f);
}

__global__ __launch_bounds__(256) void pool_out_kernel(
    const float* __restrict__ sums, const float* __restrict__ counts,
    const float* __restrict__ out_w, const float* __restrict__ out_b,
    float* __restrict__ out) {
  int i = blockIdx.x * blockDim.x + threadIdx.x;
  if (i >= NUM_GRAPHS * OUT_DIM) return;
  int g = i >> 5, o = i & (OUT_DIM - 1);
  float inv = 1.f / fmaxf(counts[g], 1.f);
  float s = 0.f;
  for (int c = 0; c < HID; ++c)
    s += sums[g * HID + c] * inv * out_w[c * OUT_DIM + o];
  out[i] = s + out_b[o];
}

// ---------------- host side ----------------
extern "C" void kernel_launch(void* const* d_in, const int* in_sizes, int n_in,
                              void* d_out, int out_size, void* d_ws, size_t ws_size,
                              hipStream_t stream) {
  const float* x        = (const float*)d_in[0];
  const float* edge_at  = (const float*)d_in[1];
  const float* in_w     = (const float*)d_in[2];
  const float* in_b     = (const float*)d_in[3];
  const float* msg_w1   = (const float*)d_in[4];
  const float* msg_b1   = (const float*)d_in[5];
  const float* msg_w2   = (const float*)d_in[6];
  const float* msg_b2   = (const float*)d_in[7];
  const float* upd_w1   = (const float*)d_in[8];
  const float* upd_b1   = (const float*)d_in[9];
  const float* upd_w2   = (const float*)d_in[10];
  const float* upd_b2   = (const float*)d_in[11];
  const float* ln_g     = (const float*)d_in[12];
  const float* ln_b     = (const float*)d_in[13];
  const float* out_w    = (const float*)d_in[14];
  const float* out_b    = (const float*)d_in[15];
  const int*   eidx     = (const int*)d_in[16];
  const int*   batch    = (const int*)d_in[17];

  // workspace carve-up (all 256B aligned)
  char* ws = (char*)d_ws;
  size_t off = 0;
  auto carve = [&](size_t bytes) -> char* {
    char* p = ws + off;
    off += (bytes + 255) & ~(size_t)255;
    return p;
  };
  float* h      = (float*)carve((size_t)N_NODES * HID * 4);
  float* aggr   = (float*)carve((size_t)N_NODES * HID * 4);
  float* sums   = (float*)carve((size_t)NUM_GRAPHS * HID * 4);  // counts adjacent
  float* counts = (float*)carve((size_t)NUM_GRAPHS * 4);
  // packed bf16 weights: tile = 32 lanes * 16 halves = 512 ushorts = 1024B
  const int T_IN  = (IN_DIM / 32)   * (HID / 16);       // 16 tiles
  const int T_M1  = (MSG_KPAD / 32) * (HID / 16);       // 72
  const int T_M2  = (HID / 32)      * (HID / 16);       // 32
  const int T_U1  = (UPD_KPAD / 32) * (HID / 16);       // 64
  const int T_U2  = T_M2;                               // 32
  unsigned short* in_wp  = (unsigned short*)carve((size_t)T_IN * 1024);
  unsigned short* m1p    = (unsigned short*)carve((size_t)N_LAYERS * T_M1 * 1024);
  unsigned short* m2p    = (unsigned short*)carve((size_t)N_LAYERS * T_M2 * 1024);
  unsigned short* u1p    = (unsigned short*)carve((size_t)N_LAYERS * T_U1 * 1024);
  unsigned short* u2p    = (unsigned short*)carve((size_t)N_LAYERS * T_U2 * 1024);
  (void)ws_size; (void)in_sizes; (void)n_in; (void)out_size;

  auto pack = [&](const float* src, int K, int N, int nKt, unsigned short* dst) {
    int thr = nKt * (N / 16) * 32;
    pack_w_kernel<<<(thr + 255) / 256, 256, 0, stream>>>(src, K, N, nKt, dst);
  };

  // 1) pack all weights
  pack(in_w, IN_DIM, HID, IN_DIM / 32, in_wp);
  for (int i = 0; i < N_LAYERS; ++i) {
    pack(msg_w1 + (size_t)i * MSG_K * HID, MSG_K, HID, MSG_KPAD / 32, m1p + (size_t)i * T_M1 * 512);
    pack(msg_w2 + (size_t)i * HID * HID,   HID,   HID, HID / 32,      m2p + (size_t)i * T_M2 * 512);
    pack(upd_w1 + (size_t)i * UPD_KPAD * HID, UPD_KPAD, HID, UPD_KPAD / 32, u1p + (size_t)i * T_U1 * 512);
    pack(upd_w2 + (size_t)i * HID * HID,   HID,   HID, HID / 32,      u2p + (size_t)i * T_U2 * 512);
  }

  // 2) input projection  (100000/32 = 3125 exact)
  in_proj_kernel<<<N_NODES / 32, 256, 0, stream>>>(x, in_wp, in_b, h);

  // 3) message-passing layers
  const int aggr_n = N_NODES * HID;
  for (int i = 0; i < N_LAYERS; ++i) {
    fill_zero_kernel<<<(aggr_n + 255) / 256, 256, 0, stream>>>(aggr, aggr_n);
    edge_msg_kernel<<<N_EDGES / 32, 256, 0, stream>>>(
        h, edge_at, eidx, eidx + N_EDGES,
        m1p + (size_t)i * T_M1 * 512, msg_b1 + (size_t)i * HID,
        m2p + (size_t)i * T_M2 * 512, msg_b2 + (size_t)i * HID, aggr);
    node_upd_kernel<<<N_NODES / 32, 256, 0, stream>>>(
        h, aggr,
        u1p + (size_t)i * T_U1 * 512, upd_b1 + (size_t)i * HID,
        u2p + (size_t)i * T_U2 * 512, upd_b2 + (size_t)i * HID,
        ln_g + (size_t)i * HID, ln_b + (size_t)i * HID, i > 0 ? 1 : 0);
  }

  // 4) global mean pool + output projection
  fill_zero_kernel<<<(NUM_GRAPHS * HID + NUM_GRAPHS + 255) / 256, 256, 0, stream>>>(
      sums, NUM_GRAPHS * HID + NUM_GRAPHS);  // counts is adjacent to sums
  pool_scatter_kernel<<<(N_NODES * HID + 255) / 256, 256, 0, stream>>>(h, batch, sums, counts);
  pool_out_kernel<<<(NUM_GRAPHS * OUT_DIM + 255) / 256, 256, 0, stream>>>(
      sums, counts, out_w, out_b, (float*)d_out);
}
